// MultiHeadAttention_3109556322805
// MI455X (gfx1250) — matrix operand order repaired and checked by reading, not compiled, not run on hardware
//
#include <hip/hip_runtime.h>
#include <stdint.h>

// MHA forward, B=4 T=1024 D=2048 H=16 DK=128.
// Compute-bound (~157 GFLOP vs ~200MB): f16 WMMA w/ f32 accum everywhere,
// flash-attention (online softmax) to avoid materializing 1024x1024 scores.

constexpr int Tc  = 1024;
constexpr int Dc  = 2048;
constexpr int DKc = 128;
constexpr int BT  = 4096;   // B*T

typedef __attribute__((ext_vector_type(16))) _Float16 v16h;
typedef __attribute__((ext_vector_type(8)))  float    v8f;

struct alignas(16) H8 { _Float16 h[8]; };
struct alignas(8)  H4 { _Float16 h[4]; };

// ---- gfx1250 async global->LDS copy (ASYNCcnt path) -----------------------
__device__ __forceinline__ void async_b128(const _Float16* gsrc, _Float16* ldst) {
  unsigned loff = (unsigned)(uintptr_t)ldst;     // low 32 bits = LDS offset
  asm volatile("global_load_async_to_lds_b128 %0, %1, off"
               :: "v"(loff), "v"((const void*)gsrc) : "memory");
}
__device__ __forceinline__ void wait_async() {
  asm volatile("s_wait_asynccnt 0x0" ::: "memory");
}

// ---- WMMA fragment loaders (ISA 7.12.2 layouts, wave32) -------------------
// A 16x32 f16: lane<16 row=lane, halves[0:8]=K0..7, [8:16]=K16..23;
//              lane>=16 same row, K8..15 / K24..31.
__device__ __forceinline__ v16h frag_a(const _Float16* base, int ldh,
                                       int row0, int k0, int lane) {
  const int r   = lane & 15;
  const int khi = (lane >> 4) << 3;
  const _Float16* p = base + (row0 + r) * ldh + k0 + khi;
  union { v16h v; H8 q[2]; } f;
  f.q[0] = *(const H8*)(p);
  f.q[1] = *(const H8*)(p + 16);
  return f.v;
}
// B 32x16 f16: lane<16 col=lane, halves[0:8]=K0..7, [8:16]=K16..23;
//              lane>=16 same col, K8..15 / K24..31 (matches frag_a).
__device__ __forceinline__ v16h frag_b(const _Float16* base, int ldh,
                                       int col0, int k0, int lane) {
  const int n    = lane & 15;
  const int koff = (lane >> 4) << 3;
  const _Float16* p = base + (col0 + n) * ldh + k0 + koff;
  union { v16h v; H8 q[2]; } f;
  f.q[0] = *(const H8*)(p);
  f.q[1] = *(const H8*)(p + 16);
  return f.v;
}

__device__ __forceinline__ v8f wmma_f16(v16h a, v16h b, v8f c) {
  return __builtin_amdgcn_wmma_f32_16x16x32_f16(false, a, false, b,
                                                (short)0, c, false, false);
}

// ---- LDS staging ----------------------------------------------------------
// 128 x 64 tile, f32 source converted to f16 on the fly. ld = 72 halves.
__device__ __forceinline__ void stage64(const float* g, long row0, long col0,
                                        int ld, _Float16* lds, int tid) {
#pragma unroll
  for (int i = 0; i < 8; ++i) {
    int ch = tid + i * 256;                  // 2048 chunks of 4 floats
    int r  = ch >> 4;
    int c  = (ch & 15) * 4;
    float4 d = *(const float4*)(g + (row0 + r) * (long)ld + col0 + c);
    H4 o4; o4.h[0] = (_Float16)d.x; o4.h[1] = (_Float16)d.y;
           o4.h[2] = (_Float16)d.z; o4.h[3] = (_Float16)d.w;
    *(H4*)(lds + r * 72 + c) = o4;
  }
}
// 128 x 64 tile, f16 source: async DMA straight into LDS.
__device__ __forceinline__ void stage64(const _Float16* g, long row0, long col0,
                                        int ld, _Float16* lds, int tid) {
#pragma unroll
  for (int i = 0; i < 4; ++i) {
    int ch = tid + i * 256;                  // 1024 chunks of 8 halves
    int r  = ch >> 3;
    int c  = (ch & 7) * 8;
    async_b128(g + (row0 + r) * (long)ld + col0 + c, lds + r * 72 + c);
  }
}
// 128 x 128 f16 tile, ld = 136 halves: async DMA into LDS.
__device__ __forceinline__ void stage128_async(const _Float16* g, int ld,
                                               _Float16* lds, int tid) {
#pragma unroll
  for (int i = 0; i < 8; ++i) {
    int ch = tid + i * 256;                  // 2048 chunks of 8 halves
    int r  = ch >> 4;
    int c  = (ch & 15) * 8;
    async_b128(g + (long)r * ld + c, lds + r * 136 + c);
  }
}
// 128 x 128 f16 tile stored TRANSPOSED (PV B-operand needs [dk][k]).
__device__ __forceinline__ void stage128T(const _Float16* g, int ld,
                                          _Float16* lds, int tid) {
#pragma unroll
  for (int i = 0; i < 8; ++i) {
    int ch = tid + i * 256;
    int r  = ch >> 4;
    int c  = (ch & 15) * 8;
    H8 d = *(const H8*)(g + (long)r * ld + c);
#pragma unroll
    for (int j = 0; j < 8; ++j) lds[(c + j) * 136 + r] = d.h[j];
  }
}

// ---- GEMM: Y[M,N] = X[M,K] @ W[N,K]^T  (W always f32 row-major) -----------
// 128x128 block tile, 8 waves, each wave 32x64 (2x4 C frags), K-step 64.
template <typename SrcT, typename OutT>
__global__ __launch_bounds__(256) void gemm_xwT(const SrcT* __restrict__ X,
                                                const float* __restrict__ W,
                                                OutT* __restrict__ Y,
                                                int M, int N, int K) {
  __shared__ alignas(16) _Float16 As[128 * 72];
  __shared__ alignas(16) _Float16 Bs[128 * 72];
  const int tid  = threadIdx.x;
  const int lane = tid & 31;
  const int wave = tid >> 5;
  const int wm   = (wave & 3) * 32;
  const int wn   = (wave >> 2) * 64;
  const long m0  = (long)blockIdx.x * 128;
  const long n0  = (long)blockIdx.y * 128;

  v8f acc[2][4];
#pragma unroll
  for (int i = 0; i < 2; ++i)
#pragma unroll
    for (int j = 0; j < 4; ++j) acc[i][j] = (v8f){};

  for (int k0 = 0; k0 < K; k0 += 64) {
    __syncthreads();
    stage64(X, m0, (long)k0, K, As, tid);
    stage64(W, n0, (long)k0, K, Bs, tid);
    if (k0 + 64 < K) {                       // gfx1250 global_prefetch_b8
      __builtin_prefetch(X + (m0 + (tid >> 1)) * (long)K + k0 + 64 + (tid & 1) * 32, 0, 0);
      __builtin_prefetch(W + (n0 + (tid >> 1)) * (long)K + k0 + 64 + (tid & 1) * 32, 0, 0);
    }
    wait_async();                            // s_wait_asynccnt 0 (no-op on f32 path)
    __syncthreads();

#pragma unroll
    for (int kk = 0; kk < 64; kk += 32) {
      v16h a[2], b[4];
#pragma unroll
      for (int mf = 0; mf < 2; ++mf) a[mf] = frag_a(As, 72, wm + mf * 16, kk, lane);
#pragma unroll
      for (int nf = 0; nf < 4; ++nf) b[nf] = frag_b(Bs, 72, wn + nf * 16, kk, lane);
#pragma unroll
      for (int mf = 0; mf < 2; ++mf)
#pragma unroll
        for (int nf = 0; nf < 4; ++nf)
          acc[mf][nf] = wmma_f16(a[mf], b[nf], acc[mf][nf]);
    }
  }

  const int rl = lane & 15;
  const int rh = (lane >> 4) * 8;
#pragma unroll
  for (int mf = 0; mf < 2; ++mf)
#pragma unroll
    for (int nf = 0; nf < 4; ++nf)
#pragma unroll
      for (int e = 0; e < 8; ++e) {
        long row = m0 + wm + mf * 16 + rh + e;
        long col = n0 + wn + nf * 16 + rl;
        Y[row * (long)N + col] = (OutT)acc[mf][nf][e];
      }
}

// ---- Flash attention: one (b,h,128-row q-tile) per workgroup --------------
// Q tile lives in LDS (staged once, async) so the compute phase keeps only
// one A-operand fragment live at a time -> stays under 256 VGPRs.
__global__ __launch_bounds__(256) void flash_attn(const _Float16* __restrict__ Q,
                                                  const _Float16* __restrict__ Km,
                                                  const _Float16* __restrict__ V,
                                                  _Float16* __restrict__ O) {
  __shared__ alignas(16) _Float16 Qt[128 * 136];
  __shared__ alignas(16) _Float16 Kt[128 * 136];
  __shared__ alignas(16) _Float16 Vt[128 * 136];   // transposed [dk][k]
  __shared__ alignas(16) _Float16 Pt[128 * 136];
  const int tid  = threadIdx.x;
  const int lane = tid & 31;
  const int wave = tid >> 5;                 // wave owns 16 query rows
  const int b    = blockIdx.y >> 4;
  const int h    = blockIdx.y & 15;
  const int q0   = blockIdx.x * 128;
  const int rl   = lane & 15;
  const int rh   = (lane >> 4) * 8;
  const long headoff = ((long)b * Tc) * Dc + (long)h * DKc;

  stage128_async(Q + headoff + (long)q0 * Dc, Dc, Qt, tid);
  wait_async();
  __syncthreads();

  v8f o[8];
  float m_i[8], l_i[8];
#pragma unroll
  for (int nf = 0; nf < 8; ++nf) o[nf] = (v8f){};
#pragma unroll
  for (int e = 0; e < 8; ++e) { m_i[e] = -1e30f; l_i[e] = 0.0f; }

  const float sc = 0.0883883476483184f;      // 1/sqrt(128)
  const int nblk = q0 / 128 + 1;             // causal: only k blocks <= q tile

  for (int kb = 0; kb < nblk; ++kb) {
    const int k0 = kb * 128;
    __syncthreads();
    stage128_async(Km + headoff + (long)k0 * Dc, Dc, Kt, tid);
    stage128T     (V  + headoff + (long)k0 * Dc, Dc, Vt, tid);
    wait_async();
    __syncthreads();

    // S = Q K^T (16x128 per wave); kf-outer keeps one Q fragment live.
    v8f s[8];
#pragma unroll
    for (int g = 0; g < 8; ++g) s[g] = (v8f){};
#pragma unroll
    for (int kf = 0; kf < 4; ++kf) {
      v16h qfrag = frag_a(Qt, 136, wave * 16, kf * 32, lane);
#pragma unroll
      for (int g = 0; g < 8; ++g) {
        v16h kfrag = frag_b(Kt, 136, g * 16, kf * 32, lane);
        s[g] = wmma_f16(qfrag, kfrag, s[g]);
      }
    }
    // scale + causal mask in f32 C layout: row=q0+wave*16+rh+e, col=k0+g*16+rl
#pragma unroll
    for (int g = 0; g < 8; ++g)
#pragma unroll
      for (int e = 0; e < 8; ++e) {
        int col = k0 + g * 16 + rl;
        int row = q0 + wave * 16 + rh + e;
        float v = s[g][e] * sc;
        s[g][e] = (col > row) ? -1e30f : v;
      }
    // online softmax: row max across 8 frags then across the 16-lane row group
    float corr[8];
#pragma unroll
    for (int e = 0; e < 8; ++e) {
      float mx = s[0][e];
#pragma unroll
      for (int g = 1; g < 8; ++g) mx = fmaxf(mx, s[g][e]);
      mx = fmaxf(mx, __shfl_xor(mx, 1, 16));
      mx = fmaxf(mx, __shfl_xor(mx, 2, 16));
      mx = fmaxf(mx, __shfl_xor(mx, 4, 16));
      mx = fmaxf(mx, __shfl_xor(mx, 8, 16));
      float mn = fmaxf(m_i[e], mx);
      corr[e]  = __expf(m_i[e] - mn);
      m_i[e]   = mn;
      l_i[e]  *= corr[e];
    }
#pragma unroll
    for (int nf = 0; nf < 8; ++nf)
#pragma unroll
      for (int e = 0; e < 8; ++e) o[nf][e] *= corr[e];

    // P = exp(S - m): accumulate row sums, spill f16 P to LDS (own rows only)
    float rs[8];
#pragma unroll
    for (int e = 0; e < 8; ++e) rs[e] = 0.0f;
#pragma unroll
    for (int g = 0; g < 8; ++g)
#pragma unroll
      for (int e = 0; e < 8; ++e) {
        float p = __expf(s[g][e] - m_i[e]);
        rs[e] += p;
        Pt[(wave * 16 + rh + e) * 136 + g * 16 + rl] = (_Float16)p;
      }
#pragma unroll
    for (int e = 0; e < 8; ++e) {
      float t = rs[e];
      t += __shfl_xor(t, 1, 16);
      t += __shfl_xor(t, 2, 16);
      t += __shfl_xor(t, 4, 16);
      t += __shfl_xor(t, 8, 16);
      l_i[e] += t;
    }

    // O += P V. kf-outer keeps a single P fragment live;
    // DS ops from the same wave are in-order, and we only read our own rows.
#pragma unroll
    for (int kf = 0; kf < 4; ++kf) {
      v16h pf = frag_a(Pt, 136, wave * 16, kf * 32, lane);
#pragma unroll
      for (int nf = 0; nf < 8; ++nf) {
        v16h vf = frag_b(Vt, 136, nf * 16, kf * 32, lane);
        o[nf] = wmma_f16(pf, vf, o[nf]);
      }
    }
  }

  // normalize and store f16 to attn workspace in [B*T, D] layout
  float inv[8];
#pragma unroll
  for (int e = 0; e < 8; ++e) inv[e] = 1.0f / l_i[e];
#pragma unroll
  for (int nf = 0; nf < 8; ++nf)
#pragma unroll
    for (int e = 0; e < 8; ++e) {
      long row = (long)b * Tc + q0 + wave * 16 + rh + e;
      long col = (long)h * DKc + nf * 16 + rl;
      O[row * (long)Dc + col] = (_Float16)(o[nf][e] * inv[e]);
    }
}

extern "C" void kernel_launch(void* const* d_in, const int* in_sizes, int n_in,
                              void* d_out, int out_size, void* d_ws, size_t ws_size,
                              hipStream_t stream) {
  (void)in_sizes; (void)n_in; (void)out_size; (void)ws_size;
  const float* query = (const float*)d_in[0];
  const float* key   = (const float*)d_in[1];
  const float* value = (const float*)d_in[2];
  // d_in[3] = mask: causal structure implemented analytically in-kernel
  const float* Wq = (const float*)d_in[4];
  const float* Wk = (const float*)d_in[5];
  const float* Wv = (const float*)d_in[6];
  const float* Wo = (const float*)d_in[7];
  float* out = (float*)d_out;

  const size_t act = (size_t)BT * Dc;        // 8.4M f16 elems per buffer
  _Float16* Qws = (_Float16*)d_ws;
  _Float16* Kws = Qws + act;
  _Float16* Vws = Kws + act;
  _Float16* Aws = Vws + act;

  dim3 blk(256);
  dim3 ggrid(BT / 128, Dc / 128);            // (32,16)

  gemm_xwT<float, _Float16><<<ggrid, blk, 0, stream>>>(query, Wq, Qws, BT, Dc, Dc);
  gemm_xwT<float, _Float16><<<ggrid, blk, 0, stream>>>(key,   Wk, Kws, BT, Dc, Dc);
  gemm_xwT<float, _Float16><<<ggrid, blk, 0, stream>>>(value, Wv, Vws, BT, Dc, Dc);

  dim3 agrid(Tc / 128, 4 * 16);              // (8, B*H)
  flash_attn<<<agrid, blk, 0, stream>>>(Qws, Kws, Vws, Aws);

  gemm_xwT<_Float16, float><<<ggrid, blk, 0, stream>>>(Aws, Wo, out, BT, Dc, Dc);
}
